// FFTConv_32615981646402
// MI455X (gfx1250) — matrix operand order repaired and checked by reading, not compiled, not run on hardware
//
#include <hip/hip_runtime.h>
#include <math.h>

#define FL     8192
#define BATCH  32
#define SEQ    262144
#define CDIM   16
#define NCOLS  128                 // output 16-col tiles per workgroup (8 waves * 16)
#define KPAD   8224                // 257 * 32 : FIR K dimension padded to 32
#define XS_LEN 10256               // 16*(NCOLS-1) + KPAD  (bf16 x window)
#define GR_LEN 8244                // reversed-filter LDS staging (+16 lead pad, +tail)

typedef __attribute__((ext_vector_type(16))) __bf16 v16bf;
typedef __attribute__((ext_vector_type(8)))  float  v8f;

union Frag16 { v16bf v; unsigned int u[8]; };

static __device__ inline unsigned short f2bf(float x) {
  union { float f; unsigned int u; } v; v.f = x;
  unsigned int r = v.u + 0x7FFFu + ((v.u >> 16) & 1u);   // round-to-nearest-even
  return (unsigned short)(r >> 16);
}
static __device__ inline unsigned int pk2(unsigned short lo, unsigned short hi) {
  return (unsigned int)lo | ((unsigned int)hi << 16);
}

// ---------------------------------------------------------------------------
// Kernel 1: per-bin frequency response H = sigmoid(c.Wf^T+bf) * exp(i(c.Wp^T+bp))
// Writes ReH / ImH as bf16 into workspace. 256K threads, trivial cost.
// ---------------------------------------------------------------------------
__global__ __launch_bounds__(256)
void k_cond(const float* __restrict__ c,
            const float* __restrict__ Wf, const float* __restrict__ bfv,
            const float* __restrict__ Wp, const float* __restrict__ bp,
            unsigned short* __restrict__ Hre, unsigned short* __restrict__ Him)
{
  int idx = blockIdx.x * 256 + threadIdx.x;
  if (idx >= BATCH * FL) return;
  int b = idx >> 13, f = idx & (FL - 1);
  float mf = bfv[f], pf = bp[f];
#pragma unroll
  for (int d = 0; d < CDIM; ++d) {
    float cv = c[b * CDIM + d];
    mf += cv * Wf[f * CDIM + d];
    pf += cv * Wp[f * CDIM + d];
  }
  float mag = 1.0f / (1.0f + __expf(-mf));
  float sn, cs;
  __sincosf(pf, &sn, &cs);
  Hre[idx] = f2bf(mag * cs);
  Him[idx] = f2bf(mag * sn);
}

// ---------------------------------------------------------------------------
// Kernel 2: g[b,k] = (1/N) * sum_f ( ReH*cos(2pi k f/N) - ImH*sin(2pi k f/N) )
// as GEMM M=8192(k) x N=32(b) x K=2*8192(f) on v_wmma_f32_16x16x32_bf16.
// Two branch-free K loops (cos half / -sin half). The trig A-matrix comes from
// a bf16 cosine LUT in LDS (no float->bf16 conversion in the loop); gathers
// are batched (load-all then pack-all) so DS waits amortize. Output is the
// REVERSED filter gr[b,c] = g[b, 8191-c] in bf16.
// ---------------------------------------------------------------------------
__global__ __launch_bounds__(256)
void k_filter(const unsigned short* __restrict__ Hre,
              const unsigned short* __restrict__ Him,
              unsigned short* __restrict__ gru)
{
  __shared__ unsigned short tabh[FL];          // bf16 cosine table, 16 KB
  int tid = threadIdx.x;
  for (int i = tid; i < FL; i += 256)
    tabh[i] = f2bf(__cosf((float)i * (6.28318530717958647692f / (float)FL)));
  __syncthreads();

  int lane = tid & 31, w = tid >> 5;
  int hh   = lane >> 4;                         // half-wave selector
  int ln   = lane & 15;                         // A row (M) / B,C column (N)
  int k0   = blockIdx.x * 64 + (w >> 1) * 16;   // M tile (filter tap rows)
  int bcol = (w & 1) * 16 + ln;                 // N column = batch index
  int krow = k0 + ln;

  // per-slot constant offsets: krow * KA, KA = hh*16 + v*2
  int krA[8];
#pragma unroll
  for (int v = 0; v < 8; ++v)
    krA[v] = krow * ((hh << 4) + (v << 1));

  const unsigned int* HreU = (const unsigned int*)Hre;
  const unsigned int* HimU = (const unsigned int*)Him;
  int bBase = bcol * (FL / 2) + (hh << 3);      // uint index base for B loads
  int kstep = krow << 5;                        // krow * 32 per K step

  v8f acc = {0.f, 0.f, 0.f, 0.f, 0.f, 0.f, 0.f, 0.f};

  int kc = 0;                                   // krow * c0 accumulator
  for (int c0 = 0; c0 < FL; c0 += 32) {         // ---- cos * ReH half ----
    Frag16 A, B;
    unsigned short tv[16];
#pragma unroll
    for (int v = 0; v < 8; ++v) {               // batched gathers, no converts
      int t = kc + krA[v];
      tv[2 * v]     = tabh[t & (FL - 1)];
      tv[2 * v + 1] = tabh[(t + krow) & (FL - 1)];
    }
#pragma unroll
    for (int v = 0; v < 8; ++v) {
      A.u[v] = pk2(tv[2 * v], tv[2 * v + 1]);
      B.u[v] = HreU[bBase + (c0 >> 1) + v];
    }
    acc = __builtin_amdgcn_wmma_f32_16x16x32_bf16(false, A.v, false, B.v,
                                                  (short)0, acc, false, false);
    kc += kstep;
  }
  kc = 2048;                                    // -sin(th) = cos(th + pi/2 ... +2048 LUT units)
  for (int c0 = 0; c0 < FL; c0 += 32) {         // ---- (-sin) * ImH half ----
    Frag16 A, B;
    unsigned short tv[16];
#pragma unroll
    for (int v = 0; v < 8; ++v) {
      int t = kc + krA[v];
      tv[2 * v]     = tabh[t & (FL - 1)];
      tv[2 * v + 1] = tabh[(t + krow) & (FL - 1)];
    }
#pragma unroll
    for (int v = 0; v < 8; ++v) {
      A.u[v] = pk2(tv[2 * v], tv[2 * v + 1]);
      B.u[v] = HimU[bBase + (c0 >> 1) + v];
    }
    acc = __builtin_amdgcn_wmma_f32_16x16x32_bf16(false, A.v, false, B.v,
                                                  (short)0, acc, false, false);
    kc += kstep;
  }

#pragma unroll
  for (int r = 0; r < 8; ++r) {                 // C layout: M = r + 8*hh, N = ln
    int k = k0 + r + (hh << 3);
    gru[bcol * FL + (FL - 1 - k)] = f2bf(acc[r] * (1.0f / (float)FL));
  }
}

// ---------------------------------------------------------------------------
// Kernel 3: per-batch 8192-tap FIR as GEMM, M=16 phases, N=16384 tiles, K=8207.
// y[16n+i] = sum_c gr[c-i] * x[16n + c - 8191].
// Workgroup = (batch, 128 columns). x window + two shifted copies of gr staged
// as bf16 in LDS; all fragment loads are aligned DS loads with immediate
// offsets off a single running base (B-side vectorizes to ds_load_b128).
// 257 wmma K-steps per wave; loop unrolled by 2.
// ---------------------------------------------------------------------------
__global__ __launch_bounds__(256)
void k_fir(const float* __restrict__ x, const unsigned short* __restrict__ gru,
           float* __restrict__ y)
{
  __shared__ unsigned short xs[XS_LEN];
  __shared__ unsigned short grsE[GR_LEN];   // grsE[i] = gr[i-16]
  __shared__ unsigned short grsO[GR_LEN];   // grsO[i] = gr[i-15]

  int b   = blockIdx.y;
  int n0  = blockIdx.x * NCOLS;
  int tid = threadIdx.x;
  const float* xb          = x + (size_t)b * SEQ;
  const unsigned short* gb = gru + b * FL;

  __builtin_prefetch(gb + ((tid << 5) & (FL - 1)), 0, 1);   // global_prefetch_b8

  int base = 16 * n0 - (FL - 1);
  for (int i2 = tid * 2; i2 < XS_LEN; i2 += 512) {
    int g0 = base + i2, g1 = g0 + 1;
    float v0 = (g0 >= 0 && g0 < SEQ) ? xb[g0] : 0.0f;
    float v1 = (g1 >= 0 && g1 < SEQ) ? xb[g1] : 0.0f;
    *(unsigned int*)&xs[i2] = pk2(f2bf(v0), f2bf(v1));
  }
  for (int i2 = tid * 2; i2 < GR_LEN; i2 += 512) {
    int j = i2 - 16;
    unsigned short e0 = (j     >= 0 && j     < FL) ? gb[j]     : (unsigned short)0;
    unsigned short e1 = (j + 1 >= 0 && j + 1 < FL) ? gb[j + 1] : (unsigned short)0;
    unsigned short o1 = (j + 2 >= 0 && j + 2 < FL) ? gb[j + 2] : (unsigned short)0;
    *(unsigned int*)&grsE[i2] = pk2(e0, e1);
    *(unsigned int*)&grsO[i2] = pk2(e1, o1);
  }
  __syncthreads();

  int lane = tid & 31, w = tid >> 5;
  int hh   = lane >> 4;
  int ln   = lane & 15;                // A row M / B,C column N
  int nl   = (w << 4) + ln;            // local output column 0..127
  const unsigned int* xsU = (const unsigned int*)xs;
  // unified index (j+16)>>1 is valid for both parities; select copy once:
  const unsigned int* gsel =
      (ln & 1) ? (const unsigned int*)grsO : (const unsigned int*)grsE;
  int aOff = ((hh << 3) + 16 - ln);    // A base numerator before >>1 (+c0)
  int bOff = (nl << 3) + (hh << 2);    // B uint base (+c0/2)

  v8f acc = {0.f, 0.f, 0.f, 0.f, 0.f, 0.f, 0.f, 0.f};
#pragma unroll 2
  for (int c0 = 0; c0 < KPAD; c0 += 32) {
    Frag16 A, B;
    int aBase = (c0 + aOff) >> 1;      // additions below are even -> floor distributes
    int bBase = bOff + (c0 >> 1);
#pragma unroll
    for (int v = 0; v < 8; ++v) {
      A.u[v] = gsel[aBase + (v & 3) + ((v >> 2) << 3)];   // +{0,1,2,3,8,9,10,11}
      B.u[v] = xsU[bBase + (v & 3) + ((v >> 2) << 3)];   // +{0,1,2,3,8,9,10,11} -> b128 pair
    }
    acc = __builtin_amdgcn_wmma_f32_16x16x32_bf16(false, A.v, false, B.v,
                                                  (short)0, acc, false, false);
  }

  // C layout: M = r + 8*hh (8 consecutive output samples), N = ln
  float* yb = y + (size_t)b * SEQ + 16 * (size_t)(n0 + nl) + (hh << 3);
#pragma unroll
  for (int r = 0; r < 8; ++r) yb[r] = acc[r];
}

// ---------------------------------------------------------------------------
extern "C" void kernel_launch(void* const* d_in, const int* in_sizes, int n_in,
                              void* d_out, int out_size, void* d_ws, size_t ws_size,
                              hipStream_t stream)
{
  (void)in_sizes; (void)n_in; (void)out_size; (void)ws_size;
  const float* x  = (const float*)d_in[0];
  const float* c  = (const float*)d_in[1];
  const float* Wf = (const float*)d_in[2];
  const float* bf = (const float*)d_in[3];
  const float* Wp = (const float*)d_in[4];
  const float* bp = (const float*)d_in[5];
  float* y = (float*)d_out;

  unsigned short* Hre = (unsigned short*)d_ws;                 // 32*8192 bf16
  unsigned short* Him = Hre + BATCH * FL;                      // 32*8192 bf16
  unsigned short* gru = Him + BATCH * FL;                      // 32*8192 bf16 (reversed filter)

  k_cond  <<<dim3((BATCH * FL) / 256), 256, 0, stream>>>(c, Wf, bf, Wp, bp, Hre, Him);
  k_filter<<<dim3(FL / 64),            256, 0, stream>>>(Hre, Him, gru);
  k_fir   <<<dim3((SEQ / 16) / NCOLS, BATCH), 256, 0, stream>>>(x, gru, y);
}
